// LSTM3lr_76690936037397
// MI455X (gfx1250) — matrix operand order repaired and checked by reading, not compiled, not run on hardware
//
#include <hip/hip_runtime.h>
#include <hip/hip_bf16.h>

// ---------------------------------------------------------------------------
// 3-layer LSTM decoder (encoder is dead code in the reference) on gfx1250.
// GEMMs run on v_wmma_f32_16x16x32_f16 with fp32 accumulation, software-
// pipelined fragment loads, and WGP-scope weight prefetch (GL2 -> WGP$).
// ---------------------------------------------------------------------------

typedef __attribute__((ext_vector_type(8)))  _Float16 v8h;
typedef __attribute__((ext_vector_type(16))) _Float16 v16h;
typedef __attribute__((ext_vector_type(8)))  float    v8f;

#define BATCH 256
#define HID   1000
#define DIN   66
#define TOUT  25
#define GPAD  1008          // per-gate padded width (63 * 16)
#define GW    (4 * GPAD)    // 4032: padded 4-gate row width
#define KH    1024          // padded K for hidden-state operand
#define KX0   96            // padded K for layer-0 input (66 -> 96)
#define CTILES 63           // GPAD / 16
#define WOUT_N 80           // padded output cols (66 -> 80)
#define LDH   1024          // row stride of h buffers (f16 elements)

union FragH { v16h v; v8h h[2]; };

__device__ __forceinline__ float sigf(float x) {
    return 1.0f / (1.0f + __expf(-x));
}

// WGP-scope prefetch: default scope on gfx1250 global_prefetch_b8 is WGP
// (fills every cache level on a miss), unlike __builtin_prefetch which
// only reaches SE/SYS scope (GL2-only, useless for L2-resident weights).
__device__ __forceinline__ void prefetch_wgp(const void* p) {
    asm volatile("global_prefetch_b8 %0, off" :: "v"(p));
}

// A fragment (16x32 f16): lane<16 -> row m0+lane, K {0..7,16..23};
// lane>=16 -> row m0+lane-16, K {8..15,24..31}. Two aligned b128 loads.
__device__ __forceinline__ void load_a(FragH& A, const _Float16* abase,
                                       int k0, int half) {
    A.h[0] = *(const v8h*)(abase + k0 +      8 * half);
    A.h[1] = *(const v8h*)(abase + k0 + 16 + 8 * half);
}

// B fragment (32x16 f16): lane -> N column n0+(lane&15), K in table order
// (lane<16: K {0..7,16..23}; lane>=16: K {8..15,24..31}).
// Load all 4 gate tiles sharing the same K rows.
__device__ __forceinline__ void load_b4(FragH* B, const _Float16* wT,
                                        int k0, int lane, int n0) {
    const _Float16* bcol = wT + (size_t)k0 * GW + n0 + (lane & 15);
    const int hf = lane >> 4;
#pragma unroll
    for (int g = 0; g < 4; ++g) {
#pragma unroll
        for (int s = 0; s < 8; ++s) {
            B[g].h[0][s] = bcol[(size_t)(8 * hf + s) * GW + g * GPAD];
            B[g].h[1][s] = bcol[(size_t)(16 + 8 * hf + s) * GW + g * GPAD];
        }
    }
}

__device__ __forceinline__ void mma4(v8f* acc, const FragH& A, const FragH* B) {
#pragma unroll
    for (int g = 0; g < 4; ++g)
        acc[g] = __builtin_amdgcn_wmma_f32_16x16x32_f16(
            false, A.v, false, B[g].v, (short)0, acc[g], false, false);
}

// ---------------------------------------------------------------------------
// Fused LSTM cell: G = X*WxT + H*WhT + b, then gates -> c,h.
// One wave owns a 16x16 tile quad (m0, n0 + g*GPAD), g=0..3; the A fragment
// is shared across the four gate WMMAs. 252 blocks x 4 waves = 1008 quads.
// Inner loops are double-buffered so loads of chunk k+1 overlap WMMAs of k.
// ---------------------------------------------------------------------------
__global__ __launch_bounds__(128)
void lstm_cell_wmma(const _Float16* __restrict__ xb, int ldx, int kx,
                    const _Float16* __restrict__ wxT,   // [kx][GW] f16
                    const _Float16* __restrict__ hb,    // [BATCH][LDH] f16
                    const _Float16* __restrict__ whT,   // [KH][GW] f16
                    const float*    __restrict__ bias,  // [4][GPAD] f32 (bih+bhh)
                    float*          __restrict__ cst,   // [BATCH][GPAD] f32 (in/out)
                    _Float16*       __restrict__ hout)  // [BATCH][LDH] f16 (out)
{
    const int lane = threadIdx.x & 31;
    const int wave = blockIdx.x * 4 + (threadIdx.x >> 5);
    const int m0   = (wave / CTILES) * 16;
    const int n0   = (wave % CTILES) * 16;
    const int half = lane >> 4;
    const int mr   = lane & 15;

    const _Float16* abase_x = xb + (size_t)(m0 + mr) * ldx;
    const _Float16* abase_h = hb + (size_t)(m0 + mr) * LDH;

    v8f acc[4] = {};

    FragH A, B[4];
    load_a(A, abase_x, 0, half);
    load_b4(B, wxT, 0, lane, n0);

    // ---- X contribution (pipelined) --------------------------------------
    for (int k0 = 0; k0 + 32 < kx; k0 += 32) {
        FragH An, Bn[4];
        load_a(An, abase_x, k0 + 32, half);
        load_b4(Bn, wxT, k0 + 32, lane, n0);
        // WGP-scope prefetch 4 chunks ahead (overrun is harmless: speculative
        // prefetches never fault and the address stays inside the workspace).
        prefetch_wgp(wxT + (size_t)(k0 + 160 + lane) * GW + n0);
        mma4(acc, A, B);
        A = An;
#pragma unroll
        for (int g = 0; g < 4; ++g) B[g] = Bn[g];
    }

    // ---- X tail overlapped with H head loads -----------------------------
    {
        FragH An, Bn[4];
        load_a(An, abase_h, 0, half);
        load_b4(Bn, whT, 0, lane, n0);
        mma4(acc, A, B);
        A = An;
#pragma unroll
        for (int g = 0; g < 4; ++g) B[g] = Bn[g];
    }

    // ---- H (recurrent) contribution (pipelined) --------------------------
    for (int k0 = 0; k0 + 32 < KH; k0 += 32) {
        FragH An, Bn[4];
        load_a(An, abase_h, k0 + 32, half);
        load_b4(Bn, whT, k0 + 32, lane, n0);
        prefetch_wgp(whT + (size_t)(k0 + 160 + lane) * GW + n0);
        mma4(acc, A, B);
        A = An;
#pragma unroll
        for (int g = 0; g < 4; ++g) B[g] = Bn[g];
    }
    mma4(acc, A, B);   // H tail

    // ---- LSTM gate epilogue ---------------------------------------------
    // C/D layout: VGPR r holds (M = m0 + r + 8*half, N = n0 + (lane&15)).
    const int  ncol  = n0 + (lane & 15);
    const int  mrow0 = m0 + 8 * half;
    const bool valid = (ncol < HID);
    const float bi = bias[0 * GPAD + ncol];
    const float bf = bias[1 * GPAD + ncol];
    const float bg = bias[2 * GPAD + ncol];
    const float bo = bias[3 * GPAD + ncol];
#pragma unroll
    for (int r = 0; r < 8; ++r) {
        const int m = mrow0 + r;
        const float gi = sigf (acc[0][r] + bi);
        const float gf = sigf (acc[1][r] + bf);
        const float gg = tanhf(acc[2][r] + bg);
        const float go = sigf (acc[3][r] + bo);
        const size_t ci = (size_t)m * GPAD + ncol;
        const float cnew = gf * cst[ci] + gi * gg;
        cst[ci] = cnew;
        const float hval = go * tanhf(cnew);
        hout[(size_t)m * LDH + ncol] = valid ? (_Float16)hval : (_Float16)0.0f;
    }
}

// ---------------------------------------------------------------------------
// Output projection: out = h2 @ Wout + bout; also feeds x for next step.
// 80 tiles (16 row-tiles x 5 col-tiles) -> 20 blocks x 4 waves. Pipelined.
// ---------------------------------------------------------------------------
__global__ __launch_bounds__(128)
void out_proj_wmma(const _Float16* __restrict__ h2,    // [BATCH][LDH]
                   const _Float16* __restrict__ woT,   // [KH][WOUT_N]
                   const float*    __restrict__ bout,  // [DIN]
                   float*          __restrict__ out,   // (BATCH, TOUT, DIN)
                   int t,
                   _Float16*       __restrict__ xnext) // [BATCH][KX0]
{
    const int lane = threadIdx.x & 31;
    const int wave = blockIdx.x * 4 + (threadIdx.x >> 5);  // 0..79
    const int m0   = (wave / 5) * 16;
    const int n0   = (wave % 5) * 16;
    const int half = lane >> 4;

    const _Float16* abase = h2 + (size_t)(m0 + (lane & 15)) * LDH;

    v8f acc = {};
    FragH A, B;
    load_a(A, abase, 0, half);
    {
        const _Float16* bcol = woT + n0 + (lane & 15);
#pragma unroll
        for (int s = 0; s < 8; ++s) {
            B.h[0][s] = bcol[(size_t)(8 * half + s) * WOUT_N];
            B.h[1][s] = bcol[(size_t)(16 + 8 * half + s) * WOUT_N];
        }
    }
    for (int k0 = 0; k0 + 32 < KH; k0 += 32) {
        FragH An, Bn;
        load_a(An, abase, k0 + 32, half);
        const _Float16* bcol = woT + (size_t)(k0 + 32) * WOUT_N + n0 + (lane & 15);
#pragma unroll
        for (int s = 0; s < 8; ++s) {
            Bn.h[0][s] = bcol[(size_t)(8 * half + s) * WOUT_N];
            Bn.h[1][s] = bcol[(size_t)(16 + 8 * half + s) * WOUT_N];
        }
        acc = __builtin_amdgcn_wmma_f32_16x16x32_f16(
            false, A.v, false, B.v, (short)0, acc, false, false);
        A = An; B = Bn;
    }
    acc = __builtin_amdgcn_wmma_f32_16x16x32_f16(
        false, A.v, false, B.v, (short)0, acc, false, false);

    const int n = n0 + (lane & 15);
    if (n < DIN) {
        const float bb = bout[n];
#pragma unroll
        for (int r = 0; r < 8; ++r) {
            const int m = m0 + 8 * half + r;
            const float v = acc[r] + bb;
            out[((size_t)m * TOUT + t) * DIN + n] = v;
            xnext[(size_t)m * KX0 + n] = (_Float16)v;
        }
    }
}

// ---------------------------------------------------------------------------
// Per-launch preparation kernels (these run on every graph replay, so they
// must be bandwidth-efficient: LDS tile transpose, coalesced both ways).
// ---------------------------------------------------------------------------

// W fp32 [4*HID][K] -> WT f16 [Kpad][GW]; WT[k][g*GPAD + j] = W[g*HID + j][k].
// Pad regions (k >= K or j >= HID) are pre-zeroed by zero_f16.
__global__ void conv_weightT_tiled(const float* __restrict__ W,
                                   _Float16* __restrict__ WT, int K)
{
    __shared__ float tile[32][33];
    const int tx = threadIdx.x, ty = threadIdx.y;
    const int rb = blockIdx.y * 32;   // source row block  (r in [0,4000))
    const int kb = blockIdx.x * 32;   // source col block  (k in [0,Kpad))

#pragma unroll
    for (int i = ty; i < 32; i += 8) {
        const int r = rb + i, k = kb + tx;
        tile[i][tx] = (r < 4 * HID && k < K) ? W[(size_t)r * K + k] : 0.0f;
    }
    __syncthreads();
#pragma unroll
    for (int i = ty; i < 32; i += 8) {
        const int k = kb + i;
        const int r = rb + tx;
        if (r < 4 * HID) {
            const int g = r / HID, j = r % HID;
            WT[(size_t)k * GW + g * GPAD + j] = (_Float16)tile[tx][i];
        }
    }
}

// Wout fp32 [HID][DIN] -> f16 [KH][WOUT_N] zero-padded (already K-major)
__global__ void conv_wout(const float* __restrict__ W, _Float16* __restrict__ WT)
{
    size_t idx = (size_t)blockIdx.x * blockDim.x + threadIdx.x;
    if (idx >= (size_t)KH * WOUT_N) return;
    int k = (int)(idx / WOUT_N), n = (int)(idx % WOUT_N);
    float v = (k < HID && n < DIN) ? W[(size_t)k * DIN + n] : 0.0f;
    WT[idx] = (_Float16)v;
}

// bias sum, padded to [4][GPAD]
__global__ void conv_bias(const float* __restrict__ bih, const float* __restrict__ bhh,
                          float* __restrict__ dst)
{
    int idx = blockIdx.x * blockDim.x + threadIdx.x;
    if (idx >= GW) return;
    int g = idx / GPAD, j = idx % GPAD;
    dst[idx] = (j < HID) ? (bih[g * HID + j] + bhh[g * HID + j]) : 0.0f;
}

__global__ void zero_f16(_Float16* p, size_t n)
{
    size_t i = (size_t)blockIdx.x * blockDim.x + threadIdx.x;
    if (i < n) p[i] = (_Float16)0.0f;
}

__global__ void zero_f32(float* p, size_t n)
{
    size_t i = (size_t)blockIdx.x * blockDim.x + threadIdx.x;
    if (i < n) p[i] = 0.0f;
}

// x0 from dec_in (B,1,D) fp32 -> f16 [BATCH][KX0] zero-padded
__global__ void init_x(const float* __restrict__ dec_in, _Float16* __restrict__ x)
{
    int idx = blockIdx.x * blockDim.x + threadIdx.x;
    if (idx >= BATCH * KX0) return;
    int m = idx / KX0, n = idx % KX0;
    x[idx] = (n < DIN) ? (_Float16)dec_in[m * DIN + n] : (_Float16)0.0f;
}

// ---------------------------------------------------------------------------
extern "C" void kernel_launch(void* const* d_in, const int* in_sizes, int n_in,
                              void* d_out, int out_size, void* d_ws, size_t ws_size,
                              hipStream_t stream)
{
    (void)in_sizes; (void)n_in; (void)out_size; (void)ws_size;

    const float* dec_in = (const float*)d_in[1];
    const float* Wih[3] = { (const float*)d_in[2],  (const float*)d_in[6],  (const float*)d_in[10] };
    const float* Whh[3] = { (const float*)d_in[3],  (const float*)d_in[7],  (const float*)d_in[11] };
    const float* bih[3] = { (const float*)d_in[4],  (const float*)d_in[8],  (const float*)d_in[12] };
    const float* bhh[3] = { (const float*)d_in[5],  (const float*)d_in[9],  (const float*)d_in[13] };
    const float* Wout = (const float*)d_in[14];
    const float* bout = (const float*)d_in[15];
    float* out = (float*)d_out;

    char* ws = (char*)d_ws;
    size_t off = 0;
    auto alloc = [&](size_t bytes) -> void* {
        void* p = ws + off;
        off = (off + bytes + 255) & ~(size_t)255;
        return p;
    };

    _Float16* WihT[3];
    _Float16* WhhT[3];
    WihT[0] = (_Float16*)alloc((size_t)KX0 * GW * sizeof(_Float16));
    WhhT[0] = (_Float16*)alloc((size_t)KH  * GW * sizeof(_Float16));
    WihT[1] = (_Float16*)alloc((size_t)KH  * GW * sizeof(_Float16));
    WhhT[1] = (_Float16*)alloc((size_t)KH  * GW * sizeof(_Float16));
    WihT[2] = (_Float16*)alloc((size_t)KH  * GW * sizeof(_Float16));
    WhhT[2] = (_Float16*)alloc((size_t)KH  * GW * sizeof(_Float16));
    _Float16* WoutT = (_Float16*)alloc((size_t)KH * WOUT_N * sizeof(_Float16));
    const size_t wtSpanElems = off / sizeof(_Float16);   // all f16 weight buffers
    float*    biasWS = (float*)alloc((size_t)3 * GW * sizeof(float));
    _Float16* hbuf   = (_Float16*)alloc((size_t)6 * BATCH * LDH * sizeof(_Float16));
    float*    cbuf   = (float*)alloc((size_t)3 * BATCH * GPAD * sizeof(float));
    _Float16* xbuf   = (_Float16*)alloc((size_t)BATCH * KX0 * sizeof(_Float16));

    _Float16* hA[3]; _Float16* hB[3]; float* cst[3];
    for (int l = 0; l < 3; ++l) {
        hA[l]  = hbuf + (size_t)l       * BATCH * LDH;
        hB[l]  = hbuf + (size_t)(3 + l) * BATCH * LDH;
        cst[l] = cbuf + (size_t)l * BATCH * GPAD;
    }

    const int THR = 256;
    auto blocks = [](size_t n) { return (int)((n + 255) / 256); };

    // ---- prep: zero all weight pads, then tiled transpose-convert --------
    zero_f16<<<blocks(wtSpanElems), THR, 0, stream>>>((_Float16*)ws, wtSpanElems);

    {
        dim3 blk(32, 8);
        dim3 g0(KX0 / 32, (4 * HID + 31) / 32);   // (3, 125)
        dim3 g1(KH  / 32, (4 * HID + 31) / 32);   // (32, 125)
        conv_weightT_tiled<<<g0, blk, 0, stream>>>(Wih[0], WihT[0], DIN);
        conv_weightT_tiled<<<g1, blk, 0, stream>>>(Whh[0], WhhT[0], HID);
        conv_weightT_tiled<<<g1, blk, 0, stream>>>(Wih[1], WihT[1], HID);
        conv_weightT_tiled<<<g1, blk, 0, stream>>>(Whh[1], WhhT[1], HID);
        conv_weightT_tiled<<<g1, blk, 0, stream>>>(Wih[2], WihT[2], HID);
        conv_weightT_tiled<<<g1, blk, 0, stream>>>(Whh[2], WhhT[2], HID);
    }
    conv_wout<<<blocks((size_t)KH * WOUT_N), THR, 0, stream>>>(Wout, WoutT);
    for (int l = 0; l < 3; ++l)
        conv_bias<<<blocks(GW), THR, 0, stream>>>(bih[l], bhh[l], biasWS + (size_t)l * GW);

    // ---- prep: zero initial states (padding columns must stay zero) ------
    zero_f16<<<blocks((size_t)6 * BATCH * LDH), THR, 0, stream>>>(hbuf, (size_t)6 * BATCH * LDH);
    zero_f32<<<blocks((size_t)3 * BATCH * GPAD), THR, 0, stream>>>(cbuf, (size_t)3 * BATCH * GPAD);
    init_x<<<blocks((size_t)BATCH * KX0), THR, 0, stream>>>(dec_in, xbuf);

    // ---- autoregressive decoder (encoder state is dead in the reference) -
    const int CELL_BLOCKS = (16 * CTILES) / 4;  // 252
    for (int t = 0; t < TOUT; ++t) {
        _Float16** hprev = (t & 1) ? hB : hA;
        _Float16** hnew  = (t & 1) ? hA : hB;

        lstm_cell_wmma<<<CELL_BLOCKS, 128, 0, stream>>>(
            xbuf, KX0, KX0, WihT[0], hprev[0], WhhT[0],
            biasWS + 0 * GW, cst[0], hnew[0]);
        lstm_cell_wmma<<<CELL_BLOCKS, 128, 0, stream>>>(
            hnew[0], LDH, KH, WihT[1], hprev[1], WhhT[1],
            biasWS + 1 * GW, cst[1], hnew[1]);
        lstm_cell_wmma<<<CELL_BLOCKS, 128, 0, stream>>>(
            hnew[1], LDH, KH, WihT[2], hprev[2], WhhT[2],
            biasWS + 2 * GW, cst[2], hnew[2]);
        out_proj_wmma<<<20, 128, 0, stream>>>(
            hnew[2], WoutT, bout, out, t, xbuf);
    }
}
